// CerberusSemanticIDBranch_62843961475556
// MI455X (gfx1250) — matrix operand order repaired and checked by reading, not compiled, not run on hardware
//
#include <hip/hip_runtime.h>
#include <hip/hip_bf16.h>

typedef __attribute__((ext_vector_type(16))) _Float16 v16h;
typedef __attribute__((ext_vector_type(8)))  _Float16 v8h;
typedef __attribute__((ext_vector_type(8)))  float    v8f;

static constexpr float kTau   = 0.07f;
static constexpr int   kBRows = 32768;
static constexpr int   kD     = 512;
static constexpr int   kPPad  = 224;   // padded proto rows (groups padded to 32)
static constexpr int   kNT    = 10;    // GEMM1 N-tiles of 16
static constexpr int   kNG    = 5;

// -------- prep kernel: pack prototypes (f16, two layouts) + column scales --------
__global__ __launch_bounds__(256) void protoprep_kernel(
    const float* __restrict__ Pg, const float* __restrict__ Ph,
    const float* __restrict__ Pt, const float* __restrict__ Pp,
    const float* __restrict__ Ps,
    _Float16* __restrict__ Pf,   // [224][512] proto-major
    _Float16* __restrict__ PT,   // [512][224] d-major
    float* __restrict__ colinv) {
  constexpr int GB32[kNG] = {0, 32, 64, 128, 192};
  constexpr int GK[kNG]   = {2, 15, 40, 40, 24};
  const int lane = threadIdx.x & 31;
  const int pr   = blockIdx.x * 8 + (threadIdx.x >> 5);
  if (pr >= kPPad) return;
  int g = 4;
  while (g > 0 && pr < GB32[g]) --g;
  const int k = pr - GB32[g];
  const bool valid = k < GK[g];
  const float* Parr[kNG] = {Pg, Ph, Pt, Pp, Ps};
  const float* sp = Parr[g] + (size_t)k * kD;
  float s = 0.0f;
  if (valid) for (int i = lane; i < kD; i += 32) { float v = sp[i]; s += v * v; }
  #pragma unroll
  for (int m = 16; m >= 1; m >>= 1) s += __shfl_xor(s, m, 32);
  const float nrm = sqrtf(s);
  for (int i = lane; i < kD; i += 32) {
    _Float16 h = valid ? (_Float16)sp[i] : (_Float16)0.0f;
    Pf[(size_t)pr * kD + i]    = h;
    PT[(size_t)i * kPPad + pr] = h;
  }
  if (lane == 0) colinv[pr] = valid ? 1.0f / ((nrm + 1e-6f) * kTau) : 0.0f;
}

// -------- fused main kernel: 1 wave = 16 rows, both GEMMs on WMMA,
//          row-norms fused into the GEMM1 load stream --------
__global__ __launch_bounds__(256) void fused_kernel(
    const float* __restrict__ x,
    const _Float16* __restrict__ Pf,
    const _Float16* __restrict__ PT,
    const float* __restrict__ colinv,
    float* __restrict__ out) {
  constexpr int NW = 8;
  __shared__ __align__(32) float    sL[NW][16][64];  // masked scaled logits
  __shared__ __align__(32) float    sS[NW][16][16];  // per-row attr marginals
  __shared__ __align__(32) _Float16 sQ[NW][16][64];  // qn staged for GEMM2 A-frags

  constexpr int TB[kNT]   = {0, 32, 64, 80, 96, 128, 144, 160, 192, 208};
  constexpr int GT0[kNG]  = {0, 1, 2, 5, 8};
  constexpr int GNT[kNG]  = {1, 1, 3, 3, 2};
  constexpr int GB32[kNG] = {0, 32, 64, 128, 192};
  constexpr int GK[kNG]   = {2, 15, 40, 40, 24};
  constexpr int GD0[kNG]  = {2, 5, 8, 8, 6};
  constexpr int GD1[kNG]  = {1, 3, 5, 5, 4};
  constexpr int GNA[kNG]  = {1, 2, 2, 2, 2};
  constexpr int GKC[kNG]  = {1, 1, 2, 2, 1};  // GEMM2 k-chunks of 32

  const int lane = threadIdx.x & 31;
  const int w    = threadIdx.x >> 5;
  const int hi   = lane >> 4;     // half-wave id
  const int ln   = lane & 15;
  const int rowbase = (blockIdx.x * NW + w) * 16;

  // ---- GEMM1: raw-x @ raw-P^T into 10 accumulator tiles (scales applied later).
  //      Per-lane sum-of-squares rides along: the two half-wave lanes with equal
  //      ln together touch each element of row (rowbase+ln) exactly once. ----
  v8f acc[kNT];
  #pragma unroll
  for (int t = 0; t < kNT; ++t)
    #pragma unroll
    for (int i = 0; i < 8; ++i) acc[t][i] = 0.0f;

  const float* xrow = x + (size_t)(rowbase + ln) * kD;
  float ssq = 0.0f;
  #pragma unroll 2
  for (int dc = 0; dc < kD / 32; ++dc) {
    const int d0 = dc * 32 + hi * 16;  // A layout: lane holds K = hi*16+e
    __builtin_prefetch(xrow + d0 + 32, 0, 1);   // next chunk (global_prefetch_b8)
    float4 f0 = *(const float4*)(xrow + d0);
    float4 f1 = *(const float4*)(xrow + d0 + 4);
    float4 f2 = *(const float4*)(xrow + d0 + 8);
    float4 f3 = *(const float4*)(xrow + d0 + 12);
    ssq += f0.x*f0.x + f0.y*f0.y + f0.z*f0.z + f0.w*f0.w;
    ssq += f1.x*f1.x + f1.y*f1.y + f1.z*f1.z + f1.w*f1.w;
    ssq += f2.x*f2.x + f2.y*f2.y + f2.z*f2.z + f2.w*f2.w;
    ssq += f3.x*f3.x + f3.y*f3.y + f3.z*f3.z + f3.w*f3.w;
    v16h a;
    a[0]  = (_Float16)f0.x; a[1]  = (_Float16)f0.y; a[2]  = (_Float16)f0.z; a[3]  = (_Float16)f0.w;
    a[4]  = (_Float16)f1.x; a[5]  = (_Float16)f1.y; a[6]  = (_Float16)f1.z; a[7]  = (_Float16)f1.w;
    a[8]  = (_Float16)f2.x; a[9]  = (_Float16)f2.y; a[10] = (_Float16)f2.z; a[11] = (_Float16)f2.w;
    a[12] = (_Float16)f3.x; a[13] = (_Float16)f3.y; a[14] = (_Float16)f3.z; a[15] = (_Float16)f3.w;
    #pragma unroll
    for (int t = 0; t < kNT; ++t) {
      // B layout: lane = proto column TB[t]+ln, contiguous K(=d): hi*16 + e
      const _Float16* bp = Pf + (size_t)(TB[t] + ln) * kD + dc * 32 + hi * 16;
      v16h b = *(const v16h*)bp;
      acc[t] = __builtin_amdgcn_wmma_f32_16x16x32_f16(false, a, false, b,
                                                      (short)0, acc[t], false, false);
    }
  }

  // fused row norms: combine the two halves, then broadcast to C-layout slots
  const float row_ssq = ssq + __shfl_xor(ssq, 16, 32);      // full row (rowbase+ln)
  const float rinv = 1.0f / (sqrtf(row_ssq) + 1e-6f);
  float rs[8];
  #pragma unroll
  for (int v = 0; v < 8; ++v) rs[v] = __shfl(rinv, v + hi * 8, 32);  // row M=v+hi*8

  // ---- per-group epilogue (softmax + factored affinity) and GEMM2 ----
  #pragma unroll
  for (int g = 0; g < kNG; ++g) {
    const int Kg = GK[g], d1 = GD1[g], na = GNA[g];
    const int kp32 = GKC[g] * 32;

    // (a) scaled, masked logits -> LDS (per-wave slab, no barriers needed)
    #pragma unroll
    for (int tt = 0; tt < GNT[g]; ++tt) {
      const int t = GT0[g] + tt;
      const int col = tt * 16 + ln;
      const float cs = colinv[GB32[g] + col];
      #pragma unroll
      for (int v = 0; v < 8; ++v) {
        float lg = acc[t][v] * rs[v] * cs;          // (x.p)/(|x||p|tau)
        sL[w][v + hi * 8][col] = (col < Kg) ? lg : -1e30f;
      }
    }

    // (b) softmax + factored affinity smoothing: lanes 0..15 own one row each
    if (hi == 0) {
      const int r = ln;
      float m = -1e30f;
      for (int j = 0; j < Kg; ++j) m = fmaxf(m, sL[w][r][j]);
      #pragma unroll
      for (int i = 0; i < 16; ++i) sS[w][r][i] = 0.0f;
      float Se = 0.0f;
      for (int j = 0; j < Kg; ++j) {
        float e = __expf(sL[w][r][j] - m);
        Se += e;
        int a0 = j / d1, a1 = j - a0 * d1;
        sS[w][r][a0] += e;                  // attr-0 marginal
        if (na == 2) sS[w][r][8 + a1] += e; // attr-1 marginal
      }
      // (qA)_j = (1/na)(s0+s1)/Se ; sum_j qA = (d0+d1)/2 exactly (or 1 for na==1)
      const float factor = (na == 2) ? 0.5f * (float)(GD0[g] + d1) : 1.0f;
      const float invna  = (na == 2) ? 0.5f : 1.0f;
      const float scale  = 1.0f / (Se * (factor + 1e-6f));
      for (int j = 0; j < kp32; ++j) {
        float qn = 0.0f;
        if (j < Kg) {
          int a0 = j / d1, a1 = j - a0 * d1;
          float s = sS[w][r][a0];
          if (na == 2) s += sS[w][r][8 + a1];
          qn = invna * s * scale;
        }
        sQ[w][r][j] = (_Float16)qn;
      }
    }

    // (c) GEMM2: qn[16 x Kpad32] @ P[Kpad32 x 512]
    v16h a2[2];
    #pragma unroll
    for (int kc = 0; kc < GKC[g]; ++kc) {
      const _Float16* qp = &sQ[w][ln][kc * 32 + hi * 16];
      v8h lo  = *(const v8h*)qp;           // K = hi*16 + 0..7
      v8h hi2 = *(const v8h*)(qp + 8);     // K = hi*16 + 8..15
      a2[kc] = __builtin_shufflevector(lo, hi2, 0, 1, 2, 3, 4, 5, 6, 7,
                                       8, 9, 10, 11, 12, 13, 14, 15);
    }
    float* og = out + (size_t)rowbase * (kNG * kD) + (size_t)g * kD;
    #pragma unroll 1
    for (int dt = 0; dt < kD / 16; ++dt) {
      v8f c;
      #pragma unroll
      for (int i = 0; i < 8; ++i) c[i] = 0.0f;
      #pragma unroll
      for (int kc = 0; kc < GKC[g]; ++kc) {
        // B layout: lane = d column dt*16+ln, contiguous K(=proto): hi*16 + e
        const _Float16* bp = PT + (size_t)(dt * 16 + ln) * kPPad
                           + GB32[g] + kc * 32 + hi * 16;
        v16h b = *(const v16h*)bp;
        c = __builtin_amdgcn_wmma_f32_16x16x32_f16(false, a2[kc], false, b,
                                                   (short)0, c, false, false);
      }
      #pragma unroll
      for (int v = 0; v < 8; ++v)
        og[(size_t)(v + hi * 8) * (kNG * kD) + dt * 16 + ln] = c[v];
    }
  }
}

extern "C" void kernel_launch(void* const* d_in, const int* in_sizes, int n_in,
                              void* d_out, int out_size, void* d_ws, size_t ws_size,
                              hipStream_t stream) {
  const float* x  = (const float*)d_in[0];
  const float* Pg = (const float*)d_in[1];
  const float* Ph = (const float*)d_in[2];
  const float* Pt = (const float*)d_in[3];
  const float* Pp = (const float*)d_in[4];
  const float* Ps = (const float*)d_in[5];
  char* ws = (char*)d_ws;
  _Float16* Pf     = (_Float16*)(ws);            // 224*512*2 = 229376 B
  _Float16* PT     = (_Float16*)(ws + 229376);   // 512*224*2 = 229376 B
  float*    colinv = (float*)(ws + 458752);      // 224*4     = 896 B
  float* out = (float*)d_out;

  protoprep_kernel<<<kPPad / 8, 256, 0, stream>>>(Pg, Ph, Pt, Pp, Ps, Pf, PT, colinv);
  fused_kernel<<<kBRows / (16 * 8), 256, 0, stream>>>(x, Pf, PT, colinv, out);
}
